// Moe_71416716198098
// MI455X (gfx1250) — compile-verified, matrix operand order checked by reading, not run on hardware
//
#include <hip/hip_runtime.h>

#define T_TOK 4096
#define HD    2048
#define ID    1408
#define NE    8
#define NP    10   // 8 routed experts + 2 shared

typedef __bf16 bf16_t;
typedef bf16_t v16bf __attribute__((ext_vector_type(16)));
typedef bf16_t v8bf  __attribute__((ext_vector_type(8)));
typedef float  v8f   __attribute__((ext_vector_type(8)));
typedef unsigned v4u __attribute__((ext_vector_type(4)));
typedef int      v8i __attribute__((ext_vector_type(8)));
typedef int      v4i __attribute__((ext_vector_type(4)));

// native RNE float->bf16 (lowers to hardware cvt on gfx1250)
__device__ __forceinline__ bf16_t f2bf(float f) { return (bf16_t)f; }
__device__ __forceinline__ unsigned short bfbits(bf16_t b) {
  return __builtin_bit_cast(unsigned short, b);
}
// two converts packed into one dword (v_cvt_pk_bf16_f32-friendly)
__device__ __forceinline__ unsigned pkbf2(float a, float b) {
  return (unsigned)bfbits((bf16_t)a) | ((unsigned)bfbits((bf16_t)b) << 16);
}
// generic LDS pointer: low 32 bits are the LDS byte offset (aperture rule)
__device__ __forceinline__ unsigned lds_off(const void* p) {
  return (unsigned)(unsigned long long)p;
}
// async global->LDS 16B copy, tracked by ASYNCcnt
__device__ __forceinline__ void async_ld16(unsigned ldsoff, const void* g) {
  asm volatile("global_load_async_to_lds_b128 %0, %1, off"
               :: "v"(ldsoff), "v"(g) : "memory");
}
__device__ __forceinline__ void wait_async0() {
  asm volatile("s_wait_asynccnt 0x0" ::: "memory");
}

// ---------------------------------------------------------------------------
// Router: one wave per token; raw logits -> d_out tail; top-2 softmax combine
// weights -> coef[t][p], shared experts (p=8,9) get 1.0.
// ---------------------------------------------------------------------------
__global__ __launch_bounds__(256) void moe_router(
    const float* __restrict__ x, const float* __restrict__ rw,
    const float* __restrict__ rb, float* __restrict__ scores_out,
    float* __restrict__ coef) {
  const int lane = threadIdx.x & 31;
  const int wave = threadIdx.x >> 5;
  const int t = blockIdx.x * 8 + wave;
  float acc[NE];
#pragma unroll
  for (int e = 0; e < NE; ++e) acc[e] = 0.f;
  const float* xrow = x + (size_t)t * HD;
  for (int h = lane; h < HD; h += 32) {
    float xv = xrow[h];
    const float4* r4 = reinterpret_cast<const float4*>(rw + (size_t)h * NE);
    float4 a = r4[0], b = r4[1];
    acc[0] += xv * a.x; acc[1] += xv * a.y; acc[2] += xv * a.z; acc[3] += xv * a.w;
    acc[4] += xv * b.x; acc[5] += xv * b.y; acc[6] += xv * b.z; acc[7] += xv * b.w;
  }
#pragma unroll
  for (int e = 0; e < NE; ++e) {
#pragma unroll
    for (int off = 16; off >= 1; off >>= 1)
      acc[e] += __shfl_down(acc[e], off, 32);
  }
  if (lane == 0) {
    float sc[NE], pe[NE];
    float mx = -3.4e38f;
#pragma unroll
    for (int e = 0; e < NE; ++e) { sc[e] = acc[e] + rb[e]; mx = fmaxf(mx, sc[e]); }
    float sum = 0.f;
#pragma unroll
    for (int e = 0; e < NE; ++e) { pe[e] = __expf(sc[e] - mx); sum += pe[e]; }
    float inv = 1.f / sum;
    int i1 = 0;
#pragma unroll
    for (int e = 1; e < NE; ++e) if (pe[e] > pe[i1]) i1 = e;
    int i2 = (i1 == 0) ? 1 : 0;
#pragma unroll
    for (int e = 0; e < NE; ++e) if (e != i1 && pe[e] > pe[i2]) i2 = e;
#pragma unroll
    for (int e = 0; e < NE; ++e) {
      scores_out[(size_t)t * NE + e] = sc[e];
      coef[(size_t)t * NP + e] = (e == i1 || e == i2) ? pe[e] * inv : 0.f;
    }
    coef[(size_t)t * NP + 8] = 1.f;
    coef[(size_t)t * NP + 9] = 1.f;
  }
}

// ---------------------------------------------------------------------------
// x (fp32) -> xb (bf16)
// ---------------------------------------------------------------------------
__global__ __launch_bounds__(256) void moe_cvt(const float* __restrict__ x,
                                               bf16_t* __restrict__ xb, int n) {
  int i = (blockIdx.x * 256 + threadIdx.x) * 4;
  if (i >= n) return;
  float4 v = *reinterpret_cast<const float4*>(x + i);
  *reinterpret_cast<uint2*>(xb + i) = make_uint2(pkbf2(v.x, v.y), pkbf2(v.z, v.w));
}

// ---------------------------------------------------------------------------
// Gate/Up GEMM + SiLU + combine-weight scaling, double-buffered LDS.
// X tile staged with global_load_async_to_lds_b128 (ASYNCcnt); fp32 weight
// tiles register-prefetched one K-step ahead, converted to bf16 into LDS.
// ---------------------------------------------------------------------------
__global__ __launch_bounds__(256) void moe_gateup(
    const bf16_t* __restrict__ xb, const float* __restrict__ eg,
    const float* __restrict__ eu, const float* __restrict__ sg,
    const float* __restrict__ su, const float* __restrict__ coef,
    bf16_t* __restrict__ act) {
  __shared__ bf16_t Xs[2][128 * 40];   // [m][k], 80B row stride: conflict-free
  __shared__ bf16_t Gs[2][64 * 40];    // [n][k]
  __shared__ bf16_t Us[2][64 * 40];

  const int tid  = threadIdx.x;
  const int lane = tid & 31;
  const int wave = tid >> 5;
  const int m0 = blockIdx.x * 128;
  const int n0 = blockIdx.y * 64;
  const int p  = blockIdx.z;

  const float* gw = (p < NE) ? (eg + (size_t)p * HD * ID)
                             : (sg + (size_t)(p - NE) * HD * ID);
  const float* uw = (p < NE) ? (eu + (size_t)p * HD * ID)
                             : (su + (size_t)(p - NE) * HD * ID);

  v8f accG[2][2] = {};
  v8f accU[2][2] = {};

  const int xrow = tid >> 1;           // 0..127
  const int xcg  = (tid & 1) * 16;     // 0 or 16 halves
  const int wn   = tid & 63;
  const int wkb  = (tid >> 6) * 8;

  const int m_off = (wave & 3) * 32;
  const int n_off = (wave >> 2) * 32;
  const int half  = lane >> 4;
  const int lmod  = lane & 15;

  // register-prefetch weights for kt = 0
  float gr[8], ur[8];
#pragma unroll
  for (int j = 0; j < 8; ++j) {
    gr[j] = gw[(size_t)(wkb + j) * ID + n0 + wn];
    ur[j] = uw[(size_t)(wkb + j) * ID + n0 + wn];
  }

  for (int kt = 0; kt < HD / 32; ++kt) {
    const int k0 = kt * 32;
    const int buf = kt & 1;
    // async-stage X tile (two 16B chunks per thread)
    {
      const bf16_t* g = xb + (size_t)(m0 + xrow) * HD + k0 + xcg;
      unsigned lo = lds_off(&Xs[buf][xrow * 40 + xcg]);
      async_ld16(lo, g);
      async_ld16(lo + 16, g + 8);
    }
    // convert + store current weight registers (packed dword LDS stores)
#pragma unroll
    for (int j = 0; j < 8; j += 2) {
      *reinterpret_cast<unsigned*>(&Gs[buf][wn * 40 + wkb + j]) = pkbf2(gr[j], gr[j + 1]);
      *reinterpret_cast<unsigned*>(&Us[buf][wn * 40 + wkb + j]) = pkbf2(ur[j], ur[j + 1]);
    }
    // register-prefetch next K-step weights (overlaps with WMMA below)
    if (kt + 1 < HD / 32) {
#pragma unroll
      for (int j = 0; j < 8; ++j) {
        gr[j] = gw[(size_t)(k0 + 32 + wkb + j) * ID + n0 + wn];
        ur[j] = uw[(size_t)(k0 + 32 + wkb + j) * ID + n0 + wn];
      }
      __builtin_prefetch(xb + (size_t)(m0 + xrow) * HD + k0 + 64 + xcg, 0, 3);
    }
    wait_async0();
    __syncthreads();

    v16bf afr[2], gfr[2], ufr[2];
#pragma unroll
    for (int mi = 0; mi < 2; ++mi) {
      const bf16_t* rp = &Xs[buf][(m_off + mi * 16 + lmod) * 40 + half * 8];
      v8bf lo = *reinterpret_cast<const v8bf*>(rp);
      v8bf hi = *reinterpret_cast<const v8bf*>(rp + 16);
      afr[mi] = __builtin_shufflevector(lo, hi, 0, 1, 2, 3, 4, 5, 6, 7,
                                        8, 9, 10, 11, 12, 13, 14, 15);
    }
#pragma unroll
    for (int ni = 0; ni < 2; ++ni) {
      const bf16_t* rg = &Gs[buf][(n_off + ni * 16 + lmod) * 40 + half * 8];
      v8bf glo = *reinterpret_cast<const v8bf*>(rg);
      v8bf ghi = *reinterpret_cast<const v8bf*>(rg + 16);
      gfr[ni] = __builtin_shufflevector(glo, ghi, 0, 1, 2, 3, 4, 5, 6, 7,
                                        8, 9, 10, 11, 12, 13, 14, 15);
      const bf16_t* ru = &Us[buf][(n_off + ni * 16 + lmod) * 40 + half * 8];
      v8bf ulo = *reinterpret_cast<const v8bf*>(ru);
      v8bf uhi = *reinterpret_cast<const v8bf*>(ru + 16);
      ufr[ni] = __builtin_shufflevector(ulo, uhi, 0, 1, 2, 3, 4, 5, 6, 7,
                                        8, 9, 10, 11, 12, 13, 14, 15);
    }
#pragma unroll
    for (int mi = 0; mi < 2; ++mi) {
#pragma unroll
      for (int ni = 0; ni < 2; ++ni) {
        accG[mi][ni] = __builtin_amdgcn_wmma_f32_16x16x32_bf16(
            false, afr[mi], false, gfr[ni], (short)0, accG[mi][ni], false, false);
        accU[mi][ni] = __builtin_amdgcn_wmma_f32_16x16x32_bf16(
            false, afr[mi], false, ufr[ni], (short)0, accU[mi][ni], false, false);
      }
    }
    // no trailing barrier: next iteration writes the other buffer
  }

  // epilogue: SiLU(g)*u, scale by combine weight, store bf16
#pragma unroll
  for (int mi = 0; mi < 2; ++mi) {
#pragma unroll
    for (int ni = 0; ni < 2; ++ni) {
#pragma unroll
      for (int r = 0; r < 8; ++r) {
        float g = accG[mi][ni][r];
        float u = accU[mi][ni][r];
        float s = g / (1.f + __expf(-g)) * u;
        int m = m0 + m_off + mi * 16 + r + half * 8;
        int n = n0 + n_off + ni * 16 + lmod;
        float c = coef[(size_t)m * NP + p];
        act[((size_t)p * T_TOK + m) * ID + n] = f2bf(s * c);
      }
    }
  }
}

// ---------------------------------------------------------------------------
// Down projection: out = x + sum_p Act'_p @ Wd_p.
// act tile staged by the Tensor Data Mover (wave 0 issues one 2D-tile
// tensor_load_to_lds per K-step, TENSORcnt); weight tiles register-prefetched.
// ---------------------------------------------------------------------------
__global__ __launch_bounds__(256) void moe_down(
    const bf16_t* __restrict__ act, const float* __restrict__ ed,
    const float* __restrict__ sd, const float* __restrict__ x,
    float* __restrict__ outp) {
  __shared__ bf16_t As[2][128 * 40];
  __shared__ bf16_t Bs[2][64 * 40];

  const int tid  = threadIdx.x;
  const int lane = tid & 31;
  const int wave = tid >> 5;
  const int m0 = blockIdx.x * 128;
  const int n0 = blockIdx.y * 64;

  const int xrow = tid >> 1;
  const int xcg  = (tid & 1) * 16;
  const int wn   = tid & 63;
  const int wkb  = (tid >> 6) * 8;

  const int m_off = (wave & 3) * 32;
  const int n_off = (wave >> 2) * 32;
  const int half  = lane >> 4;
  const int lmod  = lane & 15;

  v8f acc[2][2] = {};

  int pc = 0, ic = 0;  // current (expert-unit, I-chunk)
  float dr[8];
#pragma unroll
  for (int j = 0; j < 8; ++j)
    dr[j] = ed[(size_t)(wkb + j) * HD + n0 + wn];

  for (int q = 0; q < NP * (ID / 32); ++q) {
    const int buf = q & 1;
    const int i0 = ic * 32;

    // ---- stage act tile (128 rows x 32 halves) ----
#if __has_builtin(__builtin_amdgcn_tensor_load_to_lds)
    if (wave == 0) {
      unsigned long long ga = (unsigned long long)(const void*)(
          act + ((size_t)pc * T_TOK + m0) * ID + i0);
      unsigned lds0 = lds_off(&As[buf][0]);
      // D# group0: count=1 | lds_addr | global_addr | type=2 ("image")
      v4u g0 = { 1u, lds0, (unsigned)ga, (unsigned)(ga >> 32) | 0x80000000u };
      // D# group1: data_size=2B, pad_enable, pad_interval=16 DW, pad_amount=4 DW,
      // tensor_dim0=1408, tensor_dim1=4096, tile=32x128, stride0=1408
      v8i g1 = { 0x06D10000, (int)(1408u << 16), (int)(4096u << 16),
                 (int)(32u << 16), 128, 0, 1408, 0 };
      v4i z4 = { 0, 0, 0, 0 };
      v8i z8 = { 0, 0, 0, 0, 0, 0, 0, 0 };
      __builtin_amdgcn_tensor_load_to_lds(g0, g1, z4, z4, z8, 0);
    }
#else
    {
      const bf16_t* g = act + ((size_t)pc * T_TOK + m0 + xrow) * ID + i0 + xcg;
      unsigned lo = lds_off(&As[buf][xrow * 40 + xcg]);
      async_ld16(lo, g);
      async_ld16(lo + 16, g + 8);
    }
#endif

    // ---- store current weight registers (packed), prefetch next ----
#pragma unroll
    for (int j = 0; j < 8; j += 2)
      *reinterpret_cast<unsigned*>(&Bs[buf][wn * 40 + wkb + j]) = pkbf2(dr[j], dr[j + 1]);

    int pn = pc, in2 = ic + 1;
    if (in2 == ID / 32) { in2 = 0; ++pn; }
    if (pn < NP) {
      const float* dwn = (pn < NE) ? (ed + (size_t)pn * ID * HD)
                                   : (sd + (size_t)(pn - NE) * ID * HD);
#pragma unroll
      for (int j = 0; j < 8; ++j)
        dr[j] = dwn[(size_t)(in2 * 32 + wkb + j) * HD + n0 + wn];
    }

#if __has_builtin(__builtin_amdgcn_tensor_load_to_lds)
    if (wave == 0) __builtin_amdgcn_s_wait_tensorcnt(0);
#else
    wait_async0();
#endif
    __syncthreads();

    // ---- compute ----
    v16bf afr[2], bfr[2];
#pragma unroll
    for (int mi = 0; mi < 2; ++mi) {
      const bf16_t* rp = &As[buf][(m_off + mi * 16 + lmod) * 40 + half * 8];
      v8bf lo = *reinterpret_cast<const v8bf*>(rp);
      v8bf hi = *reinterpret_cast<const v8bf*>(rp + 16);
      afr[mi] = __builtin_shufflevector(lo, hi, 0, 1, 2, 3, 4, 5, 6, 7,
                                        8, 9, 10, 11, 12, 13, 14, 15);
    }
#pragma unroll
    for (int ni = 0; ni < 2; ++ni) {
      const bf16_t* rb_ = &Bs[buf][(n_off + ni * 16 + lmod) * 40 + half * 8];
      v8bf lo = *reinterpret_cast<const v8bf*>(rb_);
      v8bf hi = *reinterpret_cast<const v8bf*>(rb_ + 16);
      bfr[ni] = __builtin_shufflevector(lo, hi, 0, 1, 2, 3, 4, 5, 6, 7,
                                        8, 9, 10, 11, 12, 13, 14, 15);
    }
#pragma unroll
    for (int mi = 0; mi < 2; ++mi) {
#pragma unroll
      for (int ni = 0; ni < 2; ++ni) {
        acc[mi][ni] = __builtin_amdgcn_wmma_f32_16x16x32_bf16(
            false, afr[mi], false, bfr[ni], (short)0, acc[mi][ni], false, false);
      }
    }
    pc = pn; ic = in2;
  }

  // epilogue: residual add, fp32 store
#pragma unroll
  for (int mi = 0; mi < 2; ++mi) {
#pragma unroll
    for (int ni = 0; ni < 2; ++ni) {
#pragma unroll
      for (int r = 0; r < 8; ++r) {
        int m = m0 + m_off + mi * 16 + r + half * 8;
        int n = n0 + n_off + ni * 16 + lmod;
        outp[(size_t)m * HD + n] = x[(size_t)m * HD + n] + acc[mi][ni][r];
      }
    }
  }
}

// ---------------------------------------------------------------------------
extern "C" void kernel_launch(void* const* d_in, const int* in_sizes, int n_in,
                              void* d_out, int out_size, void* d_ws, size_t ws_size,
                              hipStream_t stream) {
  const float* x  = (const float*)d_in[0];
  const float* rw = (const float*)d_in[1];
  const float* rb = (const float*)d_in[2];
  const float* eg = (const float*)d_in[3];
  const float* eu = (const float*)d_in[4];
  const float* ed = (const float*)d_in[5];
  const float* sg = (const float*)d_in[6];
  const float* su = (const float*)d_in[7];
  const float* sd = (const float*)d_in[8];

  float* outp   = (float*)d_out;
  float* scores = outp + (size_t)T_TOK * HD;

  char* ws = (char*)d_ws;
  bf16_t* xb   = (bf16_t*)ws;                                            // 16 MB
  float*  coef = (float*)(ws + (size_t)T_TOK * HD * 2);                  // 160 KB
  bf16_t* act  = (bf16_t*)(ws + (size_t)T_TOK * HD * 2
                              + (size_t)T_TOK * NP * sizeof(float));     // 110 MB

  moe_router<<<T_TOK / 8, 256, 0, stream>>>(x, rw, rb, scores, coef);
  moe_cvt<<<(T_TOK * HD / 4 + 255) / 256, 256, 0, stream>>>(x, xb, T_TOK * HD);
  moe_gateup<<<dim3(T_TOK / 128, ID / 64, NP), 256, 0, stream>>>(
      xb, eg, eu, sg, su, coef, act);
  moe_down<<<dim3(T_TOK / 128, HD / 64, 1), 256, 0, stream>>>(
      act, ed, sd, x, outp);
}
